// ConvModel_506806141528
// MI455X (gfx1250) — matrix operand order repaired and checked by reading, not compiled, not run on hardware
//
#include <hip/hip_runtime.h>
#include <hip/hip_bf16.h>
#include <stdint.h>

#define VN 16000
#define EN 256000
#define BN 64
#define NTILES (EN / 32)

#define SH0f  0.28209479177387814f
#define SH1C  0.4886025119029199f
#define ISQ3  0.5773502691896258f
#define INV1  0.25f                 // 1/sqrt(16)
#define INV2  0.2041241452319315f   // 1/sqrt(24)
#define RCUT  0.25f                 // 1/CUTOFF

typedef _Float16 h2   __attribute__((ext_vector_type(2)));
typedef _Float16 v16h __attribute__((ext_vector_type(16)));
typedef float    v8f  __attribute__((ext_vector_type(8)));

union AF { v16h v; h2 h[8]; unsigned u[8]; };
union BF { v16h v; uint4 q[2]; };
union CF { v8f  v; float f[8]; };
union HF { uint4 q; h2 h[4]; };

// ---------------------------------------------------------------------------
// Bmat[u*32+kk][n] value for layer L (path-structure + normalization folded).
// ---------------------------------------------------------------------------
__device__ __forceinline__ float bmat_val(int layer, int u, int kk, int n,
                                          const float* w2, const float* b2,
                                          int numel) {
  int col = -1; float scale = 0.f;
  if (layer == 1) {                       // X = s_j[16]
    if (n < 8)        { col = u*8 + n;            scale = INV1*SH0f; }   // wg
    else if (n < 24)  { col = 128 + u*16 + (n-8); scale = INV1*SH0f; }   // ws
    else if (n < 32)  { col = 384 + u*8 + (n-24); scale = INV1;      }   // wv (sh1 applied at scatter)
  } else if (layer == 2) {                // X = [s_j(16), dot(8), vx(8), vy(8), vz(8)]
    if (n < 8) {                                                          // msg_g
      if (u < 16)       { col = u*8 + n;             scale = INV2*SH0f; } // wg0
      else if (u < 24)  { col = 128 + (u-16)*8 + n;  scale = INV2*ISQ3; } // wg1
    } else if (n < 24) {                                                  // msg_s
      if (u < 16)       { col = 192 + u*16 + (n-8);       scale = INV2*SH0f; } // ws0
      else if (u < 24)  { col = 448 + (u-16)*16 + (n-8);  scale = INV2*ISQ3; } // ws1
    } else if (n < 32) {                                                  // Ev (x sh1 at scatter)
      if (u < 16)       { col = 576 + u*8 + (n-24);  scale = INV2; }      // wv0
    } else if (n < 40) {                                                  // Fx
      if (u >= 24 && u < 32) { col = 704 + (u-24)*8 + (n-32); scale = INV2*SH0f; }
    } else if (n < 48) {                                                  // Fy
      if (u >= 32 && u < 40) { col = 704 + (u-32)*8 + (n-40); scale = INV2*SH0f; }
    } else if (n < 56) {                                                  // Fz
      if (u >= 40 && u < 48) { col = 704 + (u-40)*8 + (n-48); scale = INV2*SH0f; }
    }
  } else {                                // layer 3, X = [s_j(16), dot(8)]
    if (n < 16) {
      if (u < 16)       { col = u*16 + n;             scale = INV2*SH0f; } // w0
      else if (u < 24)  { col = 256 + (u-16)*16 + n;  scale = INV2*ISQ3; } // w1
    }
  }
  if (col < 0) return 0.f;
  if (kk < 16)  return scale * w2[kk*numel + col];
  if (kk == 16) return scale * b2[col];            // bias row (A carries X[u] at K=16)
  return 0.f;
}

// ---------------------------------------------------------------------------
// Pack B fragments in v_wmma 16x16x32 f16 B layout:
// frag[f*512 + lane*16 + i] = Bmat[K = (i<8 ? i + 8*(lane>=16) : 16 + 8*(lane>=16) + i-8)][N = nt*16 + lane%16]
// Frag counts: L1 = 32 (nt{0,1} x k0..15), L2 = 72 (t0:k0..23, t1:k0..23,
// t2:k24..39, t3:k40..47), L3 = 24.
// ---------------------------------------------------------------------------
__global__ __launch_bounds__(256) void pack_kernel(
    const float* __restrict__ w2_1, const float* __restrict__ b2_1,
    const float* __restrict__ w2_2, const float* __restrict__ b2_2,
    const float* __restrict__ w2_3, const float* __restrict__ b2_3,
    _Float16* __restrict__ f1, _Float16* __restrict__ f2,
    _Float16* __restrict__ f3) {
  int t = blockIdx.x * blockDim.x + threadIdx.x;
  int layer, idx, numel; const float *w2, *b2; _Float16* out;
  if (t < 32*512)            { layer=1; idx=t;          w2=w2_1; b2=b2_1; numel=512; out=f1; }
  else if (t < (32+72)*512)  { layer=2; idx=t-32*512;   w2=w2_2; b2=b2_2; numel=768; out=f2; }
  else if (t < 128*512)      { layer=3; idx=t-104*512;  w2=w2_3; b2=b2_3; numel=384; out=f3; }
  else return;
  int f = idx >> 9, r = idx & 511, lane = r >> 4, i = r & 15;
  int kk = (i < 8) ? (i + 8*(lane >= 16)) : (16 + 8*(lane >= 16) + (i - 8)), nl = lane & 15;
  int nt, k;
  if (layer == 1)      { nt = f >> 4; k = f & 15; }
  else if (layer == 2) { if (f < 24) { nt=0; k=f; } else if (f < 48) { nt=1; k=f-24; }
                         else if (f < 64) { nt=2; k=f-24; } else { nt=3; k=f-24; } }
  else                 { nt = 0; k = f; }
  out[idx] = (_Float16)bmat_val(layer, k, kk, nt*16 + nl, w2, b2, numel);
}

// A-fragment: P row = {X[u]*h[k] (K 0..15), X[u] (K=16), 0}
__device__ __forceinline__ void build_a(AF& a, unsigned xu, const HF& H,
                                        bool hi) {
  h2 xd = __builtin_bit_cast(h2, xu);
  a.h[0] = xd * H.h[0];
  a.h[1] = xd * H.h[1];
  a.h[2] = xd * H.h[2];
  a.h[3] = xd * H.h[3];
  a.u[4] = hi ? 0u : (xu & 0xFFFFu);
  a.u[5] = 0u; a.u[6] = 0u; a.u[7] = 0u;
}

// ---------------------------------------------------------------------------
// Fused per-edge layer kernel: radial MLP + outer-product P + WMMA GEMM +
// atomic scatter. One wave owns 32 edges (two 16-edge WMMA sub-tiles).
// ---------------------------------------------------------------------------
template <int LAYER>
__global__ __launch_bounds__(128) void edge_kernel(
    const int* __restrict__ ei, const float* __restrict__ pos,
    const float* __restrict__ freq, const float* __restrict__ w1,
    const float* __restrict__ b1, const _Float16* __restrict__ bfrag,
    const float* __restrict__ sIn, const float* __restrict__ vIn,
    float* __restrict__ gacc, float* __restrict__ sacc,
    float* __restrict__ vacc, float* __restrict__ xacc) {
  constexpr int NT = (LAYER==1) ? 2 : ((LAYER==2) ? 4 : 1);
  constexpr int NX = (LAYER==1) ? 16 : ((LAYER==2) ? 48 : 24);

  __shared__ __align__(16) _Float16 hbuf[4][32*24];   // radial hidden (f16), padded stride
  __shared__ __align__(16) unsigned xbuf[4][32*52];   // X as duplicated f16 pairs
  __shared__ int      tbuf[4][32];                    // tgt per edge
  __shared__ float    shbuf[4][32*3];                 // sh1 per edge

  const int wslot = threadIdx.x >> 5;
  const int lane  = threadIdx.x & 31;
  const bool hi   = lane >= 16;
  const int  m    = lane & 15;
  const int  wid  = blockIdx.x * 4 + wslot;
  const int  nw   = gridDim.x * 4;

  const int kbeg[4]  = {0, 0, 24, 40};
  const int kend[4]  = {(LAYER==1)?16:24, (LAYER==1)?16:24, 40, 48};
  const int fbase[4] = {0, (LAYER==1)?16:24, 48, 64};

  for (int tile = wid; tile < NTILES; tile += nw) {
    asm volatile("" ::: "memory");
    // ------------------------- Phase A: per-edge scalars -------------------
    const int  e     = tile*32 + lane;
    const bool valid = (e < EN);
    int srcv = 0, tgv = 0;
    if (valid) { srcv = ei[e]; tgv = ei[EN + e]; }
    float rx = pos[tgv*3+0] - pos[srcv*3+0];
    float ry = pos[tgv*3+1] - pos[srcv*3+1];
    float rz = pos[tgv*3+2] - pos[srcv*3+2];
    float d  = sqrtf(rx*rx + ry*ry + rz*rz);
    float ri = 1.f / fmaxf(d, 1e-9f);
    float shx = SH1C*rx*ri, shy = SH1C*ry*ri, shz = SH1C*rz*ri;
    tbuf[wslot][lane] = valid ? tgv : 0;
    shbuf[wslot][lane*3+0] = shx;
    shbuf[wslot][lane*3+1] = shy;
    shbuf[wslot][lane*3+2] = shz;

    // Bessel basis with p=6 envelope
    float xx = d * RCUT;
    float xs = fmaxf(xx, 1e-9f);
    float x2 = xs*xs, x4 = x2*x2, x5 = x4*xs, x6 = x5*xs, x7 = x6*xs;
    float env = 1.f/xs - 28.f*x5 + 48.f*x6 - 21.f*x7;
    env = (xx < 1.f) ? env : 0.f;
    float bes[10];
#pragma unroll
    for (int j = 0; j < 10; ++j) bes[j] = env * __sinf(freq[j]*xx);

    // radial hidden h = silu(bes @ w1 + b1)
    float hh[16];
#pragma unroll
    for (int k = 0; k < 16; ++k) {
      float a = b1[k];
#pragma unroll
      for (int j = 0; j < 10; ++j) a += bes[j] * w1[j*16 + k];
      hh[k] = a / (1.f + __expf(-a));
    }
#pragma unroll
    for (int k = 0; k < 8; ++k) {
      h2 p; p.x = (_Float16)hh[2*k]; p.y = (_Float16)hh[2*k+1];
      *reinterpret_cast<h2*>(&hbuf[wslot][lane*24 + 2*k]) = p;
    }

    // X vector (gathered source features)
    float Xv[NX];
#pragma unroll
    for (int u = 0; u < 16; ++u) Xv[u] = sIn[srcv*16 + u];
    if constexpr (LAYER >= 2) {
#pragma unroll
      for (int u = 0; u < 8; ++u) {
        float vx = vIn[srcv*24 + u*3 + 0];
        float vy = vIn[srcv*24 + u*3 + 1];
        float vz = vIn[srcv*24 + u*3 + 2];
        Xv[16+u] = vx*shx + vy*shy + vz*shz;          // 1o x 1o -> 0e
        if constexpr (LAYER == 2) { Xv[24+u] = vx; Xv[32+u] = vy; Xv[40+u] = vz; }
      }
    }
#pragma unroll
    for (int u = 0; u < NX; ++u) {
      float val = valid ? Xv[u] : 0.f;
      _Float16 g = (_Float16)val;
      h2 p; p.x = g; p.y = g;
      xbuf[wslot][lane*52 + u] = __builtin_bit_cast(unsigned, p);
    }
    asm volatile("" ::: "memory");

    // ------------------------- Phase B: WMMA GEMM --------------------------
#pragma unroll
    for (int st = 0; st < 2; ++st) {
      const int eloc = st*16 + m;  // A-matrix row M = lane%16
      HF H; H.q = *reinterpret_cast<const uint4*>(&hbuf[wslot][eloc*24 + (hi ? 8 : 0)]);
      int tg8[8]; float sx8[8], sy8[8], sz8[8];
#pragma unroll
      for (int r = 0; r < 8; ++r) {                 // C/D row mapping: edge = r + 8*hi
        int el = st*16 + r + (hi ? 8 : 0);
        tg8[r] = tbuf[wslot][el];
        sx8[r] = shbuf[wslot][el*3+0];
        sy8[r] = shbuf[wslot][el*3+1];
        sz8[r] = shbuf[wslot][el*3+2];
      }
#pragma unroll
      for (int nt = 0; nt < NT; ++nt) {
        CF c = {};
        const int kb = kbeg[nt], ke = kend[nt], fb = fbase[nt];
        // K-ranges are all multiples of 4: unroll x4 with independent A/B
        // register sets so A-rebuilds fill the WMMA->VALU WAR hazard window
        // instead of v_nops.
        for (int k = kb; k < ke; k += 4) {
          uint4 xq = *reinterpret_cast<const uint4*>(&xbuf[wslot][eloc*52 + k]);
          const uint4* bp = reinterpret_cast<const uint4*>(bfrag)
                          + (size_t)(fb + (k - kb)) * 64 + lane * 2;
          BF b0, b1, b2, b3;
          b0.q[0] = bp[0];   b0.q[1] = bp[1];
          b1.q[0] = bp[64];  b1.q[1] = bp[65];
          b2.q[0] = bp[128]; b2.q[1] = bp[129];
          b3.q[0] = bp[192]; b3.q[1] = bp[193];
          AF a0, a1, a2, a3;
          build_a(a0, xq.x, H, hi);
          build_a(a1, xq.y, H, hi);
          build_a(a2, xq.z, H, hi);
          build_a(a3, xq.w, H, hi);
          c.v = __builtin_amdgcn_wmma_f32_16x16x32_f16(
              false, a0.v, false, b0.v, (short)0, c.v, false, false);
          c.v = __builtin_amdgcn_wmma_f32_16x16x32_f16(
              false, a1.v, false, b1.v, (short)0, c.v, false, false);
          c.v = __builtin_amdgcn_wmma_f32_16x16x32_f16(
              false, a2.v, false, b2.v, (short)0, c.v, false, false);
          c.v = __builtin_amdgcn_wmma_f32_16x16x32_f16(
              false, a3.v, false, b3.v, (short)0, c.v, false, false);
        }
        // ----------------------- scatter (atomics) -------------------------
        if constexpr (LAYER == 3) {
#pragma unroll
          for (int r = 0; r < 8; ++r) atomicAdd(&xacc[tg8[r]*16 + m], c.f[r]);
        } else {
          if (nt == 0) {
            if (m < 8) {
#pragma unroll
              for (int r = 0; r < 8; ++r) atomicAdd(&gacc[tg8[r]*8 + m], c.f[r]);
            } else {
#pragma unroll
              for (int r = 0; r < 8; ++r) atomicAdd(&sacc[tg8[r]*16 + (m-8)], c.f[r]);
            }
          } else if (nt == 1) {
            if (m < 8) {
#pragma unroll
              for (int r = 0; r < 8; ++r) atomicAdd(&sacc[tg8[r]*16 + (m+8)], c.f[r]);
            } else {
              const int u = m - 8;                   // scalar->vector path: x sh1
#pragma unroll
              for (int r = 0; r < 8; ++r) {
                float vv = c.f[r];
                atomicAdd(&vacc[tg8[r]*24 + u*3 + 0], vv * sx8[r]);
                atomicAdd(&vacc[tg8[r]*24 + u*3 + 1], vv * sy8[r]);
                atomicAdd(&vacc[tg8[r]*24 + u*3 + 2], vv * sz8[r]);
              }
            }
          } else if (nt == 2) {                      // LAYER==2: Fx | Fy
            if (m < 8) {
#pragma unroll
              for (int r = 0; r < 8; ++r) atomicAdd(&vacc[tg8[r]*24 + m*3 + 0], c.f[r]);
            } else {
              const int u = m - 8;
#pragma unroll
              for (int r = 0; r < 8; ++r) atomicAdd(&vacc[tg8[r]*24 + u*3 + 1], c.f[r]);
            }
          } else {                                   // nt==3: Fz | pad
            if (m < 8) {
#pragma unroll
              for (int r = 0; r < 8; ++r) atomicAdd(&vacc[tg8[r]*24 + m*3 + 2], c.f[r]);
            }
          }
        }
      }
    }
  }
}

// ---------------------------------------------------------------------------
__global__ __launch_bounds__(256) void zero_kernel(float* __restrict__ p, int n) {
  int i = blockIdx.x * blockDim.x + threadIdx.x;
  if (i < n) p[i] = 0.f;
}

__global__ __launch_bounds__(256) void embed_kernel(
    const float* __restrict__ embed, const int* __restrict__ z,
    float* __restrict__ sA) {
  int i = blockIdx.x * blockDim.x + threadIdx.x;
  if (i < VN*16) sA[i] = embed[z[i >> 4]*16 + (i & 15)];
}

__global__ __launch_bounds__(256) void gate_kernel(
    const float* __restrict__ gacc, const float* __restrict__ sacc,
    const float* __restrict__ vacc, float* __restrict__ sA,
    float* __restrict__ vA) {
  int v = blockIdx.x * blockDim.x + threadIdx.x;
  if (v >= VN) return;
#pragma unroll
  for (int k = 0; k < 16; ++k) {
    float s = sacc[v*16 + k];
    sA[v*16 + k] = s / (1.f + __expf(-s));          // silu
  }
#pragma unroll
  for (int u = 0; u < 8; ++u) {
    float g = 1.f / (1.f + __expf(-gacc[v*8 + u])); // sigmoid gate
#pragma unroll
    for (int j = 0; j < 3; ++j)
      vA[v*24 + u*3 + j] = g * vacc[v*24 + u*3 + j];
  }
}

__global__ __launch_bounds__(256) void head_kernel(
    const float* __restrict__ xacc, const float* __restrict__ dw1,
    const float* __restrict__ db1, const float* __restrict__ dw2,
    const float* __restrict__ db2, const int* __restrict__ batch,
    float* __restrict__ sums, float* __restrict__ cnts) {
  int v = blockIdx.x * blockDim.x + threadIdx.x;
  if (v >= VN) return;
  float x[16];
#pragma unroll
  for (int i = 0; i < 16; ++i) x[i] = xacc[v*16 + i];
  float y = db2[0];
#pragma unroll
  for (int o = 0; o < 32; ++o) {
    float a = db1[o];
#pragma unroll
    for (int i = 0; i < 16; ++i) a += x[i] * dw1[i*32 + o];
    y += fmaxf(a, 0.f) * dw2[o];
  }
  int b = batch[v];
  atomicAdd(&sums[b], y);
  atomicAdd(&cnts[b], 1.f);
}

__global__ __launch_bounds__(64) void final_kernel(
    const float* __restrict__ sums, const float* __restrict__ cnts,
    float* __restrict__ out) {
  int b = threadIdx.x;
  if (b < BN) out[b] = sums[b] / fmaxf(cnts[b], 1.f);
}

// ---------------------------------------------------------------------------
extern "C" void kernel_launch(void* const* d_in, const int* in_sizes, int n_in,
                              void* d_out, int out_size, void* d_ws,
                              size_t ws_size, hipStream_t stream) {
  (void)in_sizes; (void)n_in; (void)out_size; (void)ws_size;
  const float* pos     = (const float*)d_in[0];
  const float* embed   = (const float*)d_in[1];
  const float* l1_freq = (const float*)d_in[2];
  const float* l1_w1   = (const float*)d_in[3];
  const float* l1_b1   = (const float*)d_in[4];
  const float* l1_w2   = (const float*)d_in[5];
  const float* l1_b2   = (const float*)d_in[6];
  const float* l2_freq = (const float*)d_in[7];
  const float* l2_w1   = (const float*)d_in[8];
  const float* l2_b1   = (const float*)d_in[9];
  const float* l2_w2   = (const float*)d_in[10];
  const float* l2_b2   = (const float*)d_in[11];
  const float* l3_freq = (const float*)d_in[12];
  const float* l3_w1   = (const float*)d_in[13];
  const float* l3_b1   = (const float*)d_in[14];
  const float* l3_w2   = (const float*)d_in[15];
  const float* l3_b2   = (const float*)d_in[16];
  const float* dw1     = (const float*)d_in[17];
  const float* db1     = (const float*)d_in[18];
  const float* dw2     = (const float*)d_in[19];
  const float* db2     = (const float*)d_in[20];
  const int*   z       = (const int*)d_in[21];
  const int*   ei      = (const int*)d_in[22];
  const int*   batch   = (const int*)d_in[23];
  float* out = (float*)d_out;

  float* ws   = (float*)d_ws;
  float* sA   = ws;                 // V*16
  float* vA   = ws + VN*16;         // V*24
  float* gacc = ws + (size_t)VN*40; // V*8
  float* sacc = ws + (size_t)VN*48; // V*16
  float* vacc = ws + (size_t)VN*64; // V*24
  float* xacc = ws + (size_t)VN*88; // V*16
  float* sums = ws + (size_t)VN*104;        // BN
  float* cnts = sums + BN;                  // BN
  _Float16* frag1 = (_Float16*)(ws + (size_t)VN*104 + 2*BN); // 32*512 halves
  _Float16* frag2 = frag1 + 32*512;                          // 72*512 halves
  _Float16* frag3 = frag2 + 72*512;                          // 24*512 halves

  pack_kernel<<<256, 256, 0, stream>>>(l1_w2, l1_b2, l2_w2, l2_b2, l3_w2, l3_b2,
                                       frag1, frag2, frag3);
  embed_kernel<<<(VN*16 + 255)/256, 256, 0, stream>>>(embed, z, sA);

  zero_kernel<<<(VN*48 + 255)/256, 256, 0, stream>>>(gacc, VN*48);
  edge_kernel<1><<<512, 128, 0, stream>>>(ei, pos, l1_freq, l1_w1, l1_b1, frag1,
                                          sA, vA, gacc, sacc, vacc, xacc);
  gate_kernel<<<(VN + 255)/256, 256, 0, stream>>>(gacc, sacc, vacc, sA, vA);

  zero_kernel<<<(VN*48 + 255)/256, 256, 0, stream>>>(gacc, VN*48);
  edge_kernel<2><<<512, 128, 0, stream>>>(ei, pos, l2_freq, l2_w1, l2_b1, frag2,
                                          sA, vA, gacc, sacc, vacc, xacc);
  gate_kernel<<<(VN + 255)/256, 256, 0, stream>>>(gacc, sacc, vacc, sA, vA);

  zero_kernel<<<(VN*16 + 2*BN + 255)/256, 256, 0, stream>>>(xacc, VN*16 + 2*BN);
  edge_kernel<3><<<512, 128, 0, stream>>>(ei, pos, l3_freq, l3_w1, l3_b1, frag3,
                                          sA, vA, gacc, sacc, vacc, xacc);

  head_kernel<<<(VN + 255)/256, 256, 0, stream>>>(xacc, dw1, db1, dw2, db2,
                                                  batch, sums, cnts);
  final_kernel<<<1, 64, 0, stream>>>(sums, cnts, out);
}
